// InvariantPointAttention_42623255446271
// MI455X (gfx1250) — matrix operand order repaired and checked by reading, not compiled, not run on hardware
//
#include <hip/hip_runtime.h>
#include <hip/hip_bf16.h>
#include <math.h>

// ---------------- problem constants ----------------
#define S_    768
#define CS_   384
#define CP_   128
#define E_    16
#define PQ_   4
#define PV_   8
#define H_    12
#define NOUT_ 2112   // H*(CP+E+PV*4)
#define QROW_ 32     // padded packed-K row for QA/KB (16 qk + 12 pts + 4 pad)
#define BTROWS_ 48   // 16 (v rows) + 24 (lvp rows) + 8 zero pad

#define SCALE_SINGLE 0.25f              // 1/sqrt(E)
#define SCALE_FRAME  (-0.11785113f)     // -1/sqrt(18*PQ) = -1/sqrt(72)

typedef __attribute__((ext_vector_type(16))) _Float16 v16h;
typedef __attribute__((ext_vector_type(8)))  float    v8f;

// ---------------- WMMA helpers (gfx1250 wave32) ----------------
__device__ __forceinline__ v8f wmma_f16(v16h a, v16h b, v8f c) {
  // D = A(16x32 f16) x B(32x16 f16) + C(16x16 f32)
  return __builtin_amdgcn_wmma_f32_16x16x32_f16(false, a, false, b, (short)0, c,
                                                false, false);
}

// A fragment from row-major f16 [M,K]: lane<16 -> row m0+lane, K 0-15; lane>=16 -> K 16-31
__device__ __forceinline__ v16h load_a_f16(const _Float16* A, long lda, int m0, int k0) {
  unsigned lane = threadIdx.x & 31u;
  int r = lane & 15, hi = lane >> 4;
  const _Float16* row = A + (long)(m0 + r) * lda + k0 + hi * 16;
  v16h a;
#pragma unroll
  for (int i = 0; i < 8; ++i) { a[i] = row[i]; a[i + 8] = row[8 + i]; }
  return a;
}

// A fragment from row-major f32 [M,K], converting to f16
__device__ __forceinline__ v16h load_a_f32cvt(const float* A, long lda, long m0, int k0) {
  unsigned lane = threadIdx.x & 31u;
  int r = lane & 15, hi = lane >> 4;
  const float* row = A + (m0 + r) * lda + k0 + hi * 16;
  v16h a;
#pragma unroll
  for (int i = 0; i < 8; ++i) { a[i] = (_Float16)row[i]; a[i + 8] = (_Float16)row[8 + i]; }
  return a;
}

// B fragment where B is stored transposed: BT[n][k] row-major, b[t]=B[k0+hi*16+t][n0+lane]
__device__ __forceinline__ v16h load_bt_f16(const _Float16* BT, long ldb, int n0, int k0) {
  unsigned lane = threadIdx.x & 31u;
  const _Float16* row = BT + (long)(n0 + (lane & 15)) * ldb + k0 + (lane >> 4) * 16;
  v16h b;
#pragma unroll
  for (int t = 0; t < 16; ++t) b[t] = row[t];
  return b;
}

// BT in f32 (weights), all 16 columns valid (no bounds branch -> no EXEC churn)
__device__ __forceinline__ v16h load_bt_f32cvt_full(const float* BT, long ldb, int n0, int k0) {
  unsigned lane = threadIdx.x & 31u;
  const float* row = BT + (long)(n0 + (lane & 15)) * ldb + k0 + (lane >> 4) * 16;
  v16h b;
#pragma unroll
  for (int t = 0; t < 16; ++t) b[t] = (_Float16)row[t];
  return b;
}

// BT in f32 with N bound (pad columns -> 0); only used for Wb (N=12)
__device__ __forceinline__ v16h load_bt_f32cvt(const float* BT, long ldb, int n0, int k0, int nmax) {
  unsigned lane = threadIdx.x & 31u;
  int n = n0 + (lane & 15);
  v16h b;
  if (n < nmax) {
    const float* row = BT + (long)n * ldb + k0 + (lane >> 4) * 16;
#pragma unroll
    for (int t = 0; t < 16; ++t) b[t] = (_Float16)row[t];
  } else {
#pragma unroll
    for (int t = 0; t < 16; ++t) b[t] = (_Float16)0.0f;
  }
  return b;
}

__device__ __forceinline__ void store_c_f32(float* C, long ldc, int m0, int n0, v8f c) {
  unsigned lane = threadIdx.x & 31u;
  int n = n0 + (lane & 15), mb = m0 + (lane >> 4) * 8;
#pragma unroll
  for (int r = 0; r < 8; ++r) C[(long)(mb + r) * ldc + n] = c[r];
}

__device__ __forceinline__ void store_c_f16(_Float16* C, long ldc, int m0, int n0, v8f c) {
  unsigned lane = threadIdx.x & 31u;
  int n = n0 + (lane & 15), mb = m0 + (lane >> 4) * 8;
#pragma unroll
  for (int r = 0; r < 8; ++r) C[(long)(mb + r) * ldc + n] = (_Float16)c[r];
}

__device__ __forceinline__ int global_wave() {
  return (int)((blockIdx.x * blockDim.x + threadIdx.x) >> 5);
}

// ---------------- kernels ----------------

// f32 -> f16 convert
__global__ void k_cvt(const float* __restrict__ src, _Float16* __restrict__ dst, long n) {
  long i = (long)blockIdx.x * blockDim.x + threadIdx.x;
  if (i < n) dst[i] = (_Float16)src[i];
}

// zero f32 buffer
__global__ void k_zero(float* __restrict__ p, long n) {
  long i = (long)blockIdx.x * blockDim.x + threadIdx.x;
  if (i < n) p[i] = 0.0f;
}

// C[M,N] = A_f16[M,K] x W_f32[N,K]^T ; output either f16 or f32
__global__ void k_proj(const _Float16* __restrict__ A, const float* __restrict__ W,
                       int M, int N, int K, _Float16* Cf16, float* Cf32) {
  int wave = global_wave();
  int ntiles = N / 16;
  int total = (M / 16) * ntiles;
  if (wave >= total) return;
  int mt = wave / ntiles, nt = wave % ntiles;
  v8f acc = {};
  for (int k0 = 0; k0 < K; k0 += 32) {
    v16h a = load_a_f16(A, K, mt * 16, k0);
    v16h b = load_bt_f32cvt_full(W, K, nt * 16, k0);
    acc = wmma_f16(a, b, acc);
  }
  if (Cf16) store_c_f16(Cf16, N, mt * 16, nt * 16, acc);
  else      store_c_f32(Cf32, N, mt * 16, nt * 16, acc);
}

// Build packed logits operands QA/KB, combined o-matrix BT, and squared norms.
__global__ void k_build(const float* __restrict__ R, const float* __restrict__ T,
                        const float* __restrict__ gamma,
                        const _Float16* __restrict__ Qh, const _Float16* __restrict__ Kh,
                        const _Float16* __restrict__ Vh,
                        const float* __restrict__ QP, const float* __restrict__ KP,
                        const float* __restrict__ VP,
                        _Float16* __restrict__ QA, _Float16* __restrict__ KB,
                        _Float16* __restrict__ BTC,
                        float* __restrict__ SQQ, float* __restrict__ SQK) {
  int idx = blockIdx.x * blockDim.x + threadIdx.x;
  if (idx >= H_ * S_) return;
  int h = idx / S_, s = idx % S_;

  float Rm[9], t3[3];
#pragma unroll
  for (int i = 0; i < 9; ++i) Rm[i] = R[s * 9 + i];
#pragma unroll
  for (int i = 0; i < 3; ++i) t3[i] = T[s * 3 + i];

  float lam = SCALE_FRAME * log1pf(expf(gamma[h]));   // scale_frame*softplus(g) < 0

  _Float16* qa = QA + ((long)h * S_ + s) * QROW_;
#pragma unroll
  for (int e = 0; e < 16; ++e) qa[e] = Qh[(long)s * (H_ * E_) + h * E_ + e];
  float sqq = 0.0f;
#pragma unroll
  for (int p = 0; p < PQ_; ++p) {
    float x0 = QP[(long)s * (H_ * PQ_ * 3) + h * 12 + p * 3 + 0];
    float x1 = QP[(long)s * (H_ * PQ_ * 3) + h * 12 + p * 3 + 1];
    float x2 = QP[(long)s * (H_ * PQ_ * 3) + h * 12 + p * 3 + 2];
#pragma unroll
    for (int d = 0; d < 3; ++d) {
      float l = Rm[d * 3 + 0] * x0 + Rm[d * 3 + 1] * x1 + Rm[d * 3 + 2] * x2 + t3[d];
      sqq += l * l;
      qa[16 + p * 3 + d] = (_Float16)l;
    }
  }
#pragma unroll
  for (int r = 28; r < 32; ++r) qa[r] = (_Float16)0.0f;
  SQQ[(long)h * S_ + s] = lam * sqq;

  _Float16* kb = KB + ((long)h * S_ + s) * QROW_;
#pragma unroll
  for (int e = 0; e < 16; ++e)
    kb[e] = (_Float16)(SCALE_SINGLE * (float)Kh[(long)s * (H_ * E_) + h * E_ + e]);
  float sqk = 0.0f;
#pragma unroll
  for (int p = 0; p < PQ_; ++p) {
    float x0 = KP[(long)s * (H_ * PQ_ * 3) + h * 12 + p * 3 + 0];
    float x1 = KP[(long)s * (H_ * PQ_ * 3) + h * 12 + p * 3 + 1];
    float x2 = KP[(long)s * (H_ * PQ_ * 3) + h * 12 + p * 3 + 2];
#pragma unroll
    for (int d = 0; d < 3; ++d) {
      float l = Rm[d * 3 + 0] * x0 + Rm[d * 3 + 1] * x1 + Rm[d * 3 + 2] * x2 + t3[d];
      sqk += l * l;
      kb[16 + p * 3 + d] = (_Float16)(-2.0f * lam * l);
    }
  }
#pragma unroll
  for (int r = 28; r < 32; ++r) kb[r] = (_Float16)0.0f;
  SQK[(long)h * S_ + s] = lam * sqk;

  _Float16* bt = BTC + (long)h * BTROWS_ * S_;
#pragma unroll
  for (int e = 0; e < 16; ++e)
    bt[(long)e * S_ + s] = Vh[(long)s * (H_ * E_) + h * E_ + e];
#pragma unroll
  for (int p = 0; p < PV_; ++p) {
    float x0 = VP[(long)s * (H_ * PV_ * 3) + h * 24 + p * 3 + 0];
    float x1 = VP[(long)s * (H_ * PV_ * 3) + h * 24 + p * 3 + 1];
    float x2 = VP[(long)s * (H_ * PV_ * 3) + h * 24 + p * 3 + 2];
#pragma unroll
    for (int d = 0; d < 3; ++d) {
      float l = Rm[d * 3 + 0] * x0 + Rm[d * 3 + 1] * x1 + Rm[d * 3 + 2] * x2 + t3[d];
      bt[(long)(16 + p * 3 + d) * S_ + s] = (_Float16)l;
    }
  }
#pragma unroll
  for (int r = 40; r < 48; ++r) bt[(long)r * S_ + s] = (_Float16)0.0f;
}

// w_pair (f16 out): WPH[h][i*S+j] = sum_c pair[i,j,c]*Wb[h,c]. M=S*S, N=16(12+pad), K=128.
__global__ void k_wpair(const float* __restrict__ pair, const float* __restrict__ Wb,
                        _Float16* __restrict__ WPH) {
  long wave = global_wave();
  long total = (long)S_ * S_ / 16;
  if (wave >= total) return;
  long m0 = wave * 16;
  v8f acc = {};
  for (int k0 = 0; k0 < CP_; k0 += 32) {
    __builtin_prefetch(pair + (m0 + 16) * CP_ + k0, 0, 1);
    v16h a = load_a_f32cvt(pair, CP_, m0, k0);
    v16h b = load_bt_f32cvt(Wb, CP_, 0, k0, H_);
    acc = wmma_f16(a, b, acc);
  }
  unsigned lane = threadIdx.x & 31u;
  int hh = lane & 15;
  long mb = m0 + (lane >> 4) * 8;
  if (hh < H_) {
#pragma unroll
    for (int r = 0; r < 8; ++r) WPH[(long)hh * S_ * S_ + mb + r] = (_Float16)acc[r];
  }
}

// Fused logits + softmax + a_sum accumulation.
__global__ void __launch_bounds__(256) k_logits_softmax(
    const _Float16* __restrict__ QA, const _Float16* __restrict__ KB,
    const float* __restrict__ SQQ, const float* __restrict__ SQK,
    const _Float16* __restrict__ WPH, _Float16* __restrict__ AH,
    float* __restrict__ ASUMF) {
  __shared__ float tile[16 * S_];   // 48 KB
  __shared__ float red[256];
  __shared__ float sinv[16];

  int blk = blockIdx.x;             // H_*(S_/16)
  int h = blk / (S_ / 16);
  int i0 = (blk % (S_ / 16)) * 16;
  int wv = (int)(threadIdx.x >> 5);
  unsigned lane = threadIdx.x & 31u;

  const _Float16* QAh = QA + (long)h * S_ * QROW_;
  const _Float16* KBh = KB + (long)h * S_ * QROW_;
  const _Float16* WPh = WPH + (long)h * S_ * S_;

  v16h a = load_a_f16(QAh, QROW_, i0, 0);       // reused across all j-tiles
  for (int jt = wv; jt < S_ / 16; jt += 8) {
    v16h b = load_bt_f16(KBh, QROW_, jt * 16, 0);
    v8f acc = {};
    acc = wmma_f16(a, b, acc);
    int j = jt * 16 + (lane & 15);
    int mb = (lane >> 4) * 8;
    float sk = SQK[(long)h * S_ + j];
#pragma unroll
    for (int r = 0; r < 8; ++r) {
      int m = mb + r;
      tile[m * S_ + j] = acc[r] + (float)WPh[(long)(i0 + m) * S_ + j]
                       + SQQ[(long)h * S_ + i0 + m] + sk;
    }
  }
  __syncthreads();

  int row = (int)(threadIdx.x >> 4);
  int tg  = (int)(threadIdx.x & 15);
  float mx = -1e30f;
  for (int j = tg; j < S_; j += 16) mx = fmaxf(mx, tile[row * S_ + j]);
  red[threadIdx.x] = mx; __syncthreads();
  for (int w = 8; w > 0; w >>= 1) {
    if (tg < w) red[threadIdx.x] = fmaxf(red[threadIdx.x], red[threadIdx.x + w]);
    __syncthreads();
  }
  mx = red[row << 4];
  __syncthreads();

  float sm = 0.0f;
  for (int j = tg; j < S_; j += 16) {
    float e = __expf(tile[row * S_ + j] - mx);
    tile[row * S_ + j] = e;
    sm += e;
  }
  red[threadIdx.x] = sm; __syncthreads();
  for (int w = 8; w > 0; w >>= 1) {
    if (tg < w) red[threadIdx.x] += red[threadIdx.x + w];
    __syncthreads();
  }
  if (tg == 0) sinv[row] = 1.0f / red[row << 4];
  __syncthreads();

  for (int idx = (int)threadIdx.x; idx < 16 * S_; idx += 256) {
    int m = idx / S_, j = idx % S_;
    AH[((long)h * S_ + (i0 + m)) * S_ + j] = (_Float16)(tile[m * S_ + j] * sinv[m]);
  }
#pragma unroll
  for (int k = 0; k < 3; ++k) {
    int c = (int)threadIdx.x + k * 256;
    float s = 0.0f;
#pragma unroll
    for (int m = 0; m < 16; ++m) s += tile[m * S_ + c] * sinv[m];
    atomicAdd(&ASUMF[(long)h * S_ + c], s);
  }
}

// a_sum f32 -> f16, padded to 16 rows
__global__ void k_asumcvt(const float* __restrict__ ASUMF, _Float16* __restrict__ ASUM) {
  int idx = blockIdx.x * blockDim.x + threadIdx.x;
  if (idx >= 16 * S_) return;
  int h = idx / S_, j = idx % S_;
  ASUM[(long)h * S_ + j] = (h < H_) ? (_Float16)ASUMF[(long)h * S_ + j] : (_Float16)0.0f;
}

// fused o_single + o_local: per head, C[S x 48] = a[h] (SxS) x BTC[h]^T.
__global__ void k_oc(const _Float16* __restrict__ AH, const _Float16* __restrict__ BTC,
                     float* __restrict__ OC) {
  int wave = global_wave();
  if (wave >= H_ * (S_ / 16)) return;
  int h = wave / (S_ / 16), mt = wave % (S_ / 16);
  const _Float16* A  = AH  + (long)h * S_ * S_;
  const _Float16* Bt = BTC + (long)h * BTROWS_ * S_;
  v8f acc0 = {}, acc1 = {}, acc2 = {};
  for (int k0 = 0; k0 < S_; k0 += 32) {
    v16h a  = load_a_f16(A, S_, mt * 16, k0);
    v16h b0 = load_bt_f16(Bt, S_, 0,  k0);
    v16h b1 = load_bt_f16(Bt, S_, 16, k0);
    v16h b2 = load_bt_f16(Bt, S_, 32, k0);
    acc0 = wmma_f16(a, b0, acc0);
    acc1 = wmma_f16(a, b1, acc1);
    acc2 = wmma_f16(a, b2, acc2);
  }
  float* C = OC + (long)h * S_ * BTROWS_;
  store_c_f32(C, BTROWS_, mt * 16, 0,  acc0);
  store_c_f32(C, BTROWS_, mt * 16, 16, acc1);
  store_c_f32(C, BTROWS_, mt * 16, 32, acc2);
}

// o_pair with CDNA5 async LDS staging: per i, C[16 x 128] = ASUM[16,S] x pair[i][S,128].
// Each wave owns 2 n-tiles (32 cols) and double-buffers 32x32 f32 chunks of pair
// into its private LDS stage with global_load_async_to_lds_b128 (ASYNCcnt path).
__global__ void __launch_bounds__(128) k_opair(const _Float16* __restrict__ ASUM,
                                               const float* __restrict__ pair,
                                               _Float16* __restrict__ ATTN) {
  __shared__ float stage[4][2][32 * 32];   // 32 KB; stage starts at LDS offset 0
  int wave = global_wave();
  if (wave >= S_ * 4) return;
  int wv = (int)(threadIdx.x >> 5);
  unsigned lane = threadIdx.x & 31u;
  int i = wave >> 2, g = wave & 3;
  int n0 = g * 32;
  const float* B = pair + (long)i * S_ * CP_ + n0;

  // staging geometry: 8 async b128 per 32x32 chunk; lane -> (row in 4-row group, 4 floats)
  int srow = (int)(lane >> 3);            // 0..3
  int scol = (int)(lane & 7) * 4;         // 0,4,...,28

  v8f acc0 = {}, acc1 = {};

  // kick chunk kt into buffer buf
#define KICK(KT, BUF)                                                            \
  {                                                                              \
    _Pragma("unroll")                                                            \
    for (int q = 0; q < 8; ++q) {                                                \
      int krow = (KT) * 32 + q * 4 + srow;                                       \
      unsigned long long ga =                                                    \
          (unsigned long long)(const void*)(B + (long)krow * CP_ + scol);        \
      unsigned la = (unsigned)(((wv * 2 + (BUF)) * 1024 +                        \
                                (q * 4 + srow) * 32 + scol) * 4);                \
      asm volatile("global_load_async_to_lds_b128 %0, %1, off"                   \
                   :: "v"(la), "v"(ga) : "memory");                              \
    }                                                                            \
  }

  KICK(0, 0);
  for (int kt = 0; kt < S_ / 32; ++kt) {
    int buf = kt & 1;
    if (kt + 1 < S_ / 32) {
      KICK(kt + 1, buf ^ 1);
      asm volatile("s_wait_asynccnt 0x8" ::: "memory");  // current chunk resident
    } else {
      asm volatile("s_wait_asynccnt 0x0" ::: "memory");
    }
    const float* ch = &stage[wv][buf][0];
    int hi = (int)(lane >> 4);
    int nn = (int)(lane & 15);
    v16h b0, b1;
#pragma unroll
    for (int t = 0; t < 16; ++t) {
      b0[t] = (_Float16)ch[(hi * 16 + t) * 32 + nn];
      b1[t] = (_Float16)ch[(hi * 16 + t) * 32 + 16 + nn];
    }
    v16h a = load_a_f16(ASUM, S_, 0, kt * 32);
    acc0 = wmma_f16(a, b0, acc0);
    acc1 = wmma_f16(a, b1, acc1);
  }
#undef KICK

  int mb = (int)(lane >> 4) * 8;
#pragma unroll
  for (int r = 0; r < 8; ++r) {
    int hh = mb + r;
    if (hh < H_) {
      int c0 = n0 + (int)(lane & 15);
      ATTN[(long)i * NOUT_ + (H_ * E_) + hh * CP_ + c0]      = (_Float16)acc0[r];
      ATTN[(long)i * NOUT_ + (H_ * E_) + hh * CP_ + c0 + 16] = (_Float16)acc1[r];
    }
  }
}

// assemble o_single / o_pts / o_norm columns of ATTN (inverse frame applied)
__global__ void k_assemble(const float* __restrict__ OC, const float* __restrict__ R,
                           const float* __restrict__ T, _Float16* __restrict__ ATTN) {
  int idx = blockIdx.x * blockDim.x + threadIdx.x;
  if (idx >= H_ * S_) return;
  int h = idx / S_, s = idx % S_;
  const float* oc = OC + ((long)h * S_ + s) * BTROWS_;
  _Float16* row = ATTN + (long)s * NOUT_;
#pragma unroll
  for (int e = 0; e < 16; ++e) row[h * E_ + e] = (_Float16)oc[e];

  float Rm[9], t3[3];
#pragma unroll
  for (int i = 0; i < 9; ++i) Rm[i] = R[s * 9 + i];
#pragma unroll
  for (int i = 0; i < 3; ++i) t3[i] = T[s * 3 + i];

  const int PTS_OFF  = H_ * E_ + H_ * CP_;     // 1728
  const int NORM_OFF = PTS_OFF + PV_ * H_ * 3; // 2016
#pragma unroll
  for (int p = 0; p < PV_; ++p) {
    float x0 = oc[16 + p * 3 + 0] - t3[0];
    float x1 = oc[16 + p * 3 + 1] - t3[1];
    float x2 = oc[16 + p * 3 + 2] - t3[2];
    float g0 = Rm[0] * x0 + Rm[3] * x1 + Rm[6] * x2;   // R^T * (x - t)
    float g1 = Rm[1] * x0 + Rm[4] * x1 + Rm[7] * x2;
    float g2 = Rm[2] * x0 + Rm[5] * x1 + Rm[8] * x2;
    row[PTS_OFF + (p * H_ + h) * 3 + 0] = (_Float16)g0;
    row[PTS_OFF + (p * H_ + h) * 3 + 1] = (_Float16)g1;
    row[PTS_OFF + (p * H_ + h) * 3 + 2] = (_Float16)g2;
    row[NORM_OFF + p * H_ + h] = (_Float16)sqrtf(g0 * g0 + g1 * g1 + g2 * g2);
  }
}

// out[S,CS] = ATTN[S,2112] x Wo[CS,2112]^T + bo; 4 n-tiles per wave (A reuse).
__global__ void k_out(const _Float16* __restrict__ ATTN, const float* __restrict__ Wo,
                      const float* __restrict__ bo, float* __restrict__ OUT) {
  int wave = global_wave();
  int ngrp = (CS_ / 16) / 4;                   // 6 groups of 4 n-tiles
  if (wave >= (S_ / 16) * ngrp) return;
  int mt = wave / ngrp, g = wave % ngrp;
  int n0 = g * 64;
  v8f acc0 = {}, acc1 = {}, acc2 = {}, acc3 = {};
  for (int k0 = 0; k0 < NOUT_; k0 += 32) {
    v16h a  = load_a_f16(ATTN, NOUT_, mt * 16, k0);
    v16h b0 = load_bt_f32cvt_full(Wo, NOUT_, n0,      k0);
    v16h b1 = load_bt_f32cvt_full(Wo, NOUT_, n0 + 16, k0);
    v16h b2 = load_bt_f32cvt_full(Wo, NOUT_, n0 + 32, k0);
    v16h b3 = load_bt_f32cvt_full(Wo, NOUT_, n0 + 48, k0);
    acc0 = wmma_f16(a, b0, acc0);
    acc1 = wmma_f16(a, b1, acc1);
    acc2 = wmma_f16(a, b2, acc2);
    acc3 = wmma_f16(a, b3, acc3);
  }
  unsigned lane = threadIdx.x & 31u;
  int mb = mt * 16 + (lane >> 4) * 8;
  int nn = (int)(lane & 15);
#pragma unroll
  for (int t = 0; t < 4; ++t) {
    int n = n0 + t * 16 + nn;
    float bb = bo[n];
    v8f* accp = (t == 0) ? &acc0 : (t == 1) ? &acc1 : (t == 2) ? &acc2 : &acc3;
#pragma unroll
    for (int r = 0; r < 8; ++r) OUT[(long)(mb + r) * CS_ + n] = (*accp)[r] + bb;
  }
}

// ---------------- host launch ----------------
extern "C" void kernel_launch(void* const* d_in, const int* in_sizes, int n_in,
                              void* d_out, int out_size, void* d_ws, size_t ws_size,
                              hipStream_t stream) {
  const float* single = (const float*)d_in[0];
  const float* pair   = (const float*)d_in[1];
  const float* rot    = (const float*)d_in[2];
  const float* trans  = (const float*)d_in[3];
  const float* Wq     = (const float*)d_in[4];
  const float* Wk     = (const float*)d_in[5];
  const float* Wv     = (const float*)d_in[6];
  const float* Wqp    = (const float*)d_in[7];
  const float* Wkp    = (const float*)d_in[8];
  const float* Wvp    = (const float*)d_in[9];
  const float* Wb     = (const float*)d_in[10];
  const float* Wo     = (const float*)d_in[11];
  const float* bo     = (const float*)d_in[12];
  const float* gamma  = (const float*)d_in[13];
  float* out = (float*)d_out;

  size_t off = 0;
  auto carve = [&](size_t bytes) -> char* {
    off = (off + 255) & ~(size_t)255;
    char* p = (char*)d_ws + off;
    off += bytes;
    return p;
  };
  _Float16* SINGLE_H = (_Float16*)carve((size_t)S_ * CS_ * 2);
  _Float16* Qh    = (_Float16*)carve((size_t)S_ * H_ * E_ * 2);
  _Float16* Kh    = (_Float16*)carve((size_t)S_ * H_ * E_ * 2);
  _Float16* Vh    = (_Float16*)carve((size_t)S_ * H_ * E_ * 2);
  float*    QPf   = (float*)carve((size_t)S_ * H_ * PQ_ * 3 * 4);
  float*    KPf   = (float*)carve((size_t)S_ * H_ * PQ_ * 3 * 4);
  float*    VPf   = (float*)carve((size_t)S_ * H_ * PV_ * 3 * 4);
  _Float16* QA    = (_Float16*)carve((size_t)H_ * S_ * QROW_ * 2);
  _Float16* KB    = (_Float16*)carve((size_t)H_ * S_ * QROW_ * 2);
  _Float16* BTC   = (_Float16*)carve((size_t)H_ * BTROWS_ * S_ * 2);
  float*    SQQ   = (float*)carve((size_t)H_ * S_ * 4);
  float*    SQK   = (float*)carve((size_t)H_ * S_ * 4);
  _Float16* WPH   = (_Float16*)carve((size_t)H_ * S_ * S_ * 2);
  _Float16* AH    = (_Float16*)carve((size_t)H_ * S_ * S_ * 2);
  float*    ASUMF = (float*)carve((size_t)H_ * S_ * 4);
  _Float16* ASUM  = (_Float16*)carve((size_t)16 * S_ * 2);
  float*    OC    = (float*)carve((size_t)H_ * S_ * BTROWS_ * 4);
  _Float16* ATTN  = (_Float16*)carve((size_t)S_ * NOUT_ * 2);
  (void)ws_size; (void)in_sizes; (void)n_in; (void)out_size;

  k_cvt<<<(S_ * CS_ + 255) / 256, 256, 0, stream>>>(single, SINGLE_H, (long)S_ * CS_);
  k_zero<<<(H_ * S_ + 255) / 256, 256, 0, stream>>>(ASUMF, (long)H_ * S_);

  k_proj<<<(48 * 12) / 4, 128, 0, stream>>>(SINGLE_H, Wq, S_, H_ * E_, CS_, Qh, nullptr);
  k_proj<<<(48 * 12) / 4, 128, 0, stream>>>(SINGLE_H, Wk, S_, H_ * E_, CS_, Kh, nullptr);
  k_proj<<<(48 * 12) / 4, 128, 0, stream>>>(SINGLE_H, Wv, S_, H_ * E_, CS_, Vh, nullptr);
  k_proj<<<(48 * 9)  / 4, 128, 0, stream>>>(SINGLE_H, Wqp, S_, H_ * PQ_ * 3, CS_, nullptr, QPf);
  k_proj<<<(48 * 9)  / 4, 128, 0, stream>>>(SINGLE_H, Wkp, S_, H_ * PQ_ * 3, CS_, nullptr, KPf);
  k_proj<<<(48 * 18) / 4, 128, 0, stream>>>(SINGLE_H, Wvp, S_, H_ * PV_ * 3, CS_, nullptr, VPf);

  k_build<<<(H_ * S_ + 255) / 256, 256, 0, stream>>>(rot, trans, gamma, Qh, Kh, Vh,
                                                     QPf, KPf, VPf, QA, KB, BTC, SQQ, SQK);

  k_wpair<<<((S_ * S_ / 16) + 7) / 8, 256, 0, stream>>>(pair, Wb, WPH);

  k_logits_softmax<<<H_ * (S_ / 16), 256, 0, stream>>>(QA, KB, SQQ, SQK, WPH, AH, ASUMF);

  k_asumcvt<<<(16 * S_ + 255) / 256, 256, 0, stream>>>(ASUMF, ASUM);

  k_oc<<<(H_ * 48) / 4, 128, 0, stream>>>(AH, BTC, OC);

  k_opair<<<(S_ * 4) / 4, 128, 0, stream>>>(ASUM, pair, ATTN);

  k_assemble<<<(H_ * S_ + 255) / 256, 256, 0, stream>>>(OC, rot, trans, ATTN);

  k_out<<<(48 * 6) / 4, 128, 0, stream>>>(ATTN, Wo, bo, out);
}
